// MultiHeadSelfAttention_17351667876403
// MI455X (gfx1250) — matrix operand order repaired and checked by reading, not compiled, not run on hardware
//
#include <hip/hip_runtime.h>
#include <cmath>

// ---------------------------------------------------------------------------
// MI455X (gfx1250) fused multi-head self-attention with ALiBi + adjacency bias
//   x(2,2048,1024) -> QKV gemm -> flash-attn (bf16 WMMA, f32 accum) -> out gemm
// WMMA: v_wmma_f32_16x16x32_bf16; K/V double-buffered in LDS via Tensor Data
// Mover (TENSORcnt), DMA overlapped with compute.
// ---------------------------------------------------------------------------

typedef __bf16 bf16_t;
typedef __attribute__((ext_vector_type(16))) __bf16 v16bf;
typedef __attribute__((ext_vector_type(8)))  __bf16 v8bf;
typedef __attribute__((ext_vector_type(8)))  float  v8f;
typedef __attribute__((ext_vector_type(4)))  unsigned int v4u;
typedef __attribute__((ext_vector_type(8)))  int v8i;
typedef __attribute__((ext_vector_type(4)))  int v4i;

#if defined(__has_builtin)
#if __has_builtin(__builtin_amdgcn_tensor_load_to_lds)
#define HAVE_TDM 1
#endif
#endif

#define WMMA_BF16(a, b, c) \
  __builtin_amdgcn_wmma_f32_16x16x32_bf16(false, (a), false, (b), (short)0, (c), false, false)

__device__ __forceinline__ unsigned short f2bf(float f) {
  unsigned int u = __builtin_bit_cast(unsigned int, f);
  u += 0x7FFFu + ((u >> 16) & 1u);           // round-to-nearest-even
  return (unsigned short)(u >> 16);
}

__device__ __forceinline__ v16bf cat8(v8bf lo, v8bf hi) {
  return __builtin_shufflevector(lo, hi, 0,1,2,3,4,5,6,7,8,9,10,11,12,13,14,15);
}

// A-fragment (16x32 bf16): lane l holds row M=l%16; element j at
// K=16*(l>>4)+j  -> two adjacent 16B chunks.
__device__ __forceinline__ v16bf load_a_frag(const unsigned short* row, int hi) {
  v8bf lo  = *(const v8bf*)(row + 16 * hi);
  v8bf hi8 = *(const v8bf*)(row + 16 * hi + 8);
  return cat8(lo, hi8);
}

__device__ __forceinline__ float rmax16(float v) {
  v = fmaxf(v, __shfl_xor(v, 1, 32));
  v = fmaxf(v, __shfl_xor(v, 2, 32));
  v = fmaxf(v, __shfl_xor(v, 4, 32));
  v = fmaxf(v, __shfl_xor(v, 8, 32));
  return v;
}
__device__ __forceinline__ float rsum16(float v) {
  v += __shfl_xor(v, 1, 32);
  v += __shfl_xor(v, 2, 32);
  v += __shfl_xor(v, 4, 32);
  v += __shfl_xor(v, 8, 32);
  return v;
}

#ifdef HAVE_TDM
// TDM 2-D tile load: D# per CDNA5 ISA ch.8 (count=1, type=2 "image",
// data_size=2B, tile0 x tile1 with row stride; groups 2/3 unused -> zeros).
__device__ __forceinline__ void tdm_load_2d(unsigned lds_off, const void* gptr,
                                            unsigned tile0, unsigned tile1,
                                            unsigned long long stride0) {
  unsigned long long ga = (unsigned long long)(uintptr_t)gptr;
  unsigned long long lo = 1ull /*count=1*/ | ((unsigned long long)lds_off << 32);
  unsigned long long hi = (ga & 0x01FFFFFFFFFFFFFFull) | (2ull << 62) /*type=2*/;
  v4u g0 = { (unsigned)lo, (unsigned)(lo >> 32), (unsigned)hi, (unsigned)(hi >> 32) };
  // group1: data_size=1(2B)@16; tensor_dim0@48(32b); tensor_dim1@80(32b);
  // tile_dim0@112(16b); tile_dim1@128(16b); tensor_dim0_stride@160(48b)
  unsigned long long q0 = (1ull << 16) | ((unsigned long long)(tile0 & 0xFFFFu) << 48);
  unsigned long long q1 = ((unsigned long long)tile1 << 16)      // tensor_dim1 = tile1
                        | ((unsigned long long)(tile0 & 0xFFFFu) << 48); // tile_dim0
  unsigned long long q2 = (unsigned long long)(tile1 & 0xFFFFu)  // tile_dim1
                        | ((stride0 & 0xFFFFFFFFull) << 32);
  unsigned long long q3 = (stride0 >> 32) & 0xFFFFull;
  v8i g1 = { (int)q0, (int)(q0 >> 32), (int)q1, (int)(q1 >> 32),
             (int)q2, (int)(q2 >> 32), (int)q3, (int)(q3 >> 32) };
  v4i z4 = { 0, 0, 0, 0 };
#if __clang_major__ >= 23
  v8i z8 = { 0, 0, 0, 0, 0, 0, 0, 0 };
  __builtin_amdgcn_tensor_load_to_lds(g0, g1, z4, z4, z8, 0);
#else
  __builtin_amdgcn_tensor_load_to_lds(g0, g1, z4, z4, 0);
#endif
}
#endif

static constexpr int Bn = 2, Ln = 2048, Dn = 1024, Hn = 16, HSn = 64, TDn = 3072;
static constexpr int Mn = Bn * Ln; // 4096 rows

// ---------------- conversion kernels -------------------------------------
__global__ void cvt_f32_bf16(const float* __restrict__ s, unsigned short* __restrict__ d, int n) {
  int i = blockIdx.x * 256 + threadIdx.x;
  if (i < n) d[i] = f2bf(s[i]);
}

__global__ void transpose_cvt(const float* __restrict__ s, unsigned short* __restrict__ d,
                              int rows, int cols) {
  int i = blockIdx.x * 256 + threadIdx.x;
  if (i >= rows * cols) return;
  int c = i / rows, r = i - c * rows;
  d[i] = f2bf(s[(size_t)r * cols + c]);
}

// ---------------- QKV projection GEMM (32x32 per wave, 2x2 WMMA) ----------
__global__ void qkv_gemm(const unsigned short* __restrict__ xb,
                         const unsigned short* __restrict__ wtb,
                         const float* __restrict__ in_bias,
                         unsigned short* __restrict__ Qb,
                         unsigned short* __restrict__ Kb,
                         unsigned short* __restrict__ Vt) {
  const int lane = threadIdx.x & 31, wid = threadIdx.x >> 5;
  const int ln = lane & 15, hi = lane >> 4;
  const int tile = blockIdx.x * 8 + wid;       // 12288 tiles of 32x32
  const int NT = TDn / 32;                     // 96
  const int mt = tile / NT, nt = tile % NT;

  const unsigned short* arow0 = xb + (size_t)(mt * 32 + ln) * Dn;
  const unsigned short* arow1 = arow0 + 16 * Dn;
  const unsigned short* brow0 = wtb + (size_t)(nt * 32 + ln) * Dn;
  const unsigned short* brow1 = brow0 + 16 * Dn;

  v8f c[2][2] = {};
  for (int kb = 0; kb < Dn; kb += 32) {
    __builtin_prefetch(arow0 + kb + 128, 0, 1);
    v16bf a0 = load_a_frag(arow0 + kb, hi);
    v16bf a1 = load_a_frag(arow1 + kb, hi);
    v16bf b0 = *(const v16bf*)(brow0 + kb + 16 * hi);
    v16bf b1 = *(const v16bf*)(brow1 + kb + 16 * hi);
    c[0][0] = WMMA_BF16(a0, b0, c[0][0]);
    c[0][1] = WMMA_BF16(a0, b1, c[0][1]);
    c[1][0] = WMMA_BF16(a1, b0, c[1][0]);
    c[1][1] = WMMA_BF16(a1, b1, c[1][1]);
  }

#pragma unroll
  for (int v = 0; v < 2; ++v) {
    const int colN = nt * 32 + v * 16 + ln;
    const int h = colN / 192, r3 = colN % 192;
    const int sel = r3 / 64, hs = r3 % 64;
    const float bias = in_bias[colN];
#pragma unroll
    for (int u = 0; u < 2; ++u) {
#pragma unroll
      for (int i = 0; i < 8; ++i) {
        int m = mt * 32 + u * 16 + i + 8 * hi;
        int bb = m >> 11, lr = m & (Ln - 1);
        unsigned short val = f2bf(c[u][v][i] + bias);
        if (sel == 0)      Qb[(((size_t)bb * Hn + h) * Ln + lr) * HSn + hs] = val;
        else if (sel == 1) Kb[(((size_t)bb * Hn + h) * Ln + lr) * HSn + hs] = val;
        else               Vt[(((size_t)bb * Hn + h) * HSn + hs) * Ln + lr] = val;
      }
    }
  }
}

// ---------------- flash attention -----------------------------------------
// Workgroup = 8 waves sharing one (b,h); each wave owns a 16-row query block.
// K/V 32-wide blocks double-buffered in LDS via TDM; DMA overlaps compute.
__global__ void attn_kernel(const unsigned short* __restrict__ Qb,
                            const unsigned short* __restrict__ Kb,
                            const unsigned short* __restrict__ Vt,
                            const float* __restrict__ adj,
                            const unsigned char* __restrict__ msk,
                            const float* __restrict__ gamma,
                            unsigned short* __restrict__ Ob) {
  const int lane = threadIdx.x & 31, wid = threadIdx.x >> 5;
  const int ln = lane & 15, hi = lane >> 4;
  const int w = blockIdx.x * 8 + wid;          // 4096 waves
  const int mblk = w & 127, bh = w >> 7;       // 8 waves/block share bh
  const int h = bh & 15, b = bh >> 4;
  const int qbase = mblk * 16;
  constexpr int NBLK = Ln / 32;                // 64 KV blocks

  const unsigned short* Qp = Qb + ((size_t)(b * Hn + h) * Ln) * HSn;
  const unsigned short* Kp = Kb + ((size_t)(b * Hn + h) * Ln) * HSn;
  const unsigned short* Vp = Vt + ((size_t)(b * Hn + h) * HSn) * Ln;
  const float* adjp = adj + (size_t)b * Ln * Ln;
  const unsigned char* mp = msk + (size_t)b * Ln * Ln;
  const float g = gamma[h];
  const float slope = (h < 8) ? exp2f(-(float)(h + 1)) : 0.0f;
  const float NEGBIG = -3.402823466e38f;

  __shared__ __align__(16) unsigned short Kls[2][32 * 64]; // 2 x (32 kv x 64 hs)
  __shared__ __align__(16) unsigned short Vls[2][64 * 32]; // 2 x (64 hs x 32 kv)
  __shared__ __align__(16) unsigned short pbuf[8][16 * 32];
  unsigned short* pw = &pbuf[wid][0];

  const unsigned short* qrow = Qp + (size_t)(qbase + ln) * HSn;
  v16bf aq0 = load_a_frag(qrow, hi);
  v16bf aq1 = load_a_frag(qrow + 32, hi);

  float m8[8], l8[8];
  v8f o[4] = {};
#pragma unroll
  for (int i = 0; i < 8; ++i) { m8[i] = NEGBIG; l8[i] = 0.0f; }

#ifdef HAVE_TDM
  const unsigned koff0 = (unsigned)(uintptr_t)(void*)&Kls[0][0];
  const unsigned koff1 = (unsigned)(uintptr_t)(void*)&Kls[1][0];
  const unsigned voff0 = (unsigned)(uintptr_t)(void*)&Vls[0][0];
  const unsigned voff1 = (unsigned)(uintptr_t)(void*)&Vls[1][0];
  if (wid == 0) {  // prologue: stream block 0 into buffer 0
    tdm_load_2d(koff0, Kp, 64, 32, 64);
    tdm_load_2d(voff0, Vp, 32, 64, Ln);
  }
#endif

  for (int blk = 0; blk < NBLK; ++blk) {
    const int kv = blk * 32;
    const int cur = blk & 1;
#ifdef HAVE_TDM
    __syncthreads(); // all readers of spare buffer (iter blk-1) are done
    if (wid == 0) {
      if (blk + 1 < NBLK) {
        // prefetch next block into spare buffer, overlapping with compute
        const unsigned kn = cur ? koff0 : koff1;
        const unsigned vn = cur ? voff0 : voff1;
        tdm_load_2d(kn, Kp + (size_t)(kv + 32) * HSn, 64, 32, 64);
        tdm_load_2d(vn, Vp + (kv + 32),               32, 64, Ln);
        // in-order TDM completion: <=2 outstanding => current buffer ready
        __builtin_amdgcn_s_wait_tensorcnt(2);
      } else {
        __builtin_amdgcn_s_wait_tensorcnt(0);
      }
    }
    __syncthreads(); // publish current buffer to all waves
#else
    { // cooperative fallback: 256 lanes x 16B each for K and V blocks
      int t = threadIdx.x;
      *(uint4*)(&Kls[cur][0] + t * 8) = *(const uint4*)(Kp + (size_t)kv * HSn + t * 8);
      int vr = (t * 8) >> 5, vc = (t * 8) & 31;
      *(uint4*)(&Vls[cur][0] + t * 8) = *(const uint4*)(Vp + (size_t)vr * Ln + kv + vc);
    }
    __syncthreads();
#endif
    const unsigned short* Kc = &Kls[cur][0];
    const unsigned short* Vc = &Vls[cur][0];

    // ---- S = Q K^T from LDS-staged K
    v8f s0 = {}, s1 = {};
    {
      const unsigned short* kr0 = Kc + (size_t)ln * HSn;
      s0 = WMMA_BF16(aq0, *(const v16bf*)(kr0 + 16 * hi), s0);
      s0 = WMMA_BF16(aq1, *(const v16bf*)(kr0 + 32 + 16 * hi), s0);
      const unsigned short* kr1 = Kc + (size_t)(16 + ln) * HSn;
      s1 = WMMA_BF16(aq0, *(const v16bf*)(kr1 + 16 * hi), s1);
      s1 = WMMA_BF16(aq1, *(const v16bf*)(kr1 + 32 + 16 * hi), s1);
    }

    // ---- bias + mask + online softmax
#pragma unroll
    for (int i = 0; i < 8; ++i) {
      const int q = qbase + i + 8 * hi;
      const size_t ro = (size_t)q * Ln;
      const int c0 = kv + ln, c1 = kv + 16 + ln;
      float a0 = s0[i] * 0.125f - slope * fabsf((float)(q - c0)) + g * adjp[ro + c0];
      float a1 = s1[i] * 0.125f - slope * fabsf((float)(q - c1)) + g * adjp[ro + c1];
      if (!mp[ro + c0]) a0 = NEGBIG;
      if (!mp[ro + c1]) a1 = NEGBIG;
      float mnew = fmaxf(m8[i], rmax16(fmaxf(a0, a1)));
      float corr = __expf(m8[i] - mnew);
      float p0 = __expf(a0 - mnew), p1 = __expf(a1 - mnew);
      l8[i] = l8[i] * corr + rsum16(p0 + p1);
      m8[i] = mnew;
      o[0][i] *= corr; o[1][i] *= corr; o[2][i] *= corr; o[3][i] *= corr;
      pw[(i + 8 * hi) * 32 + ln]      = f2bf(p0);
      pw[(i + 8 * hi) * 32 + 16 + ln] = f2bf(p1);
    }
    __asm__ volatile("" ::: "memory"); // same-wave DS order (DScnt in-order)

    // ---- O += P V from LDS
    const unsigned short* pr = pw + ln * 32;
    v16bf pa = cat8(*(const v8bf*)(pr + 16 * hi), *(const v8bf*)(pr + 16 * hi + 8));
#pragma unroll
    for (int t = 0; t < 4; ++t) {
      v16bf vb = *(const v16bf*)(Vc + (size_t)(t * 16 + ln) * 32 + 16 * hi);
      o[t] = WMMA_BF16(pa, vb, o[t]);
    }
#ifndef HAVE_TDM
    __syncthreads(); // all readers done before next block's staging overwrites
#endif
  }

#pragma unroll
  for (int t = 0; t < 4; ++t) {
#pragma unroll
    for (int i = 0; i < 8; ++i) {
      int q = qbase + i + 8 * hi;
      float v = o[t][i] / l8[i];
      Ob[((size_t)(b * Ln + q)) * Dn + h * HSn + t * 16 + ln] = f2bf(v);
    }
  }
}

// ---------------- output projection GEMM (32x32 per wave) -----------------
__global__ void out_gemm(const unsigned short* __restrict__ Ob,
                         const unsigned short* __restrict__ otb,
                         const float* __restrict__ out_bias,
                         float* __restrict__ out) {
  const int lane = threadIdx.x & 31, wid = threadIdx.x >> 5;
  const int ln = lane & 15, hi = lane >> 4;
  const int tile = blockIdx.x * 8 + wid;       // 4096 tiles of 32x32
  const int NT = Dn / 32;                      // 32
  const int mt = tile / NT, nt = tile % NT;

  const unsigned short* arow0 = Ob + (size_t)(mt * 32 + ln) * Dn;
  const unsigned short* arow1 = arow0 + 16 * Dn;
  const unsigned short* brow0 = otb + (size_t)(nt * 32 + ln) * Dn;
  const unsigned short* brow1 = brow0 + 16 * Dn;

  v8f c[2][2] = {};
  for (int kb = 0; kb < Dn; kb += 32) {
    __builtin_prefetch(arow0 + kb + 128, 0, 1);
    v16bf a0 = load_a_frag(arow0 + kb, hi);
    v16bf a1 = load_a_frag(arow1 + kb, hi);
    v16bf b0 = *(const v16bf*)(brow0 + kb + 16 * hi);
    v16bf b1 = *(const v16bf*)(brow1 + kb + 16 * hi);
    c[0][0] = WMMA_BF16(a0, b0, c[0][0]);
    c[0][1] = WMMA_BF16(a0, b1, c[0][1]);
    c[1][0] = WMMA_BF16(a1, b0, c[1][0]);
    c[1][1] = WMMA_BF16(a1, b1, c[1][1]);
  }
#pragma unroll
  for (int v = 0; v < 2; ++v) {
    const int n = nt * 32 + v * 16 + ln;
    const float bias = out_bias[n];
#pragma unroll
    for (int u = 0; u < 2; ++u) {
#pragma unroll
      for (int i = 0; i < 8; ++i) {
        int m = mt * 32 + u * 16 + i + 8 * hi;
        out[(size_t)m * Dn + n] = c[u][v][i] + bias;
      }
    }
  }
}

// ---------------- launch ---------------------------------------------------
extern "C" void kernel_launch(void* const* d_in, const int* in_sizes, int n_in,
                              void* d_out, int out_size, void* d_ws, size_t ws_size,
                              hipStream_t stream) {
  const float* x        = (const float*)d_in[0];
  const float* adj      = (const float*)d_in[1];
  const float* weights  = (const float*)d_in[2];
  const float* in_bias  = (const float*)d_in[3];
  const float* out_w    = (const float*)d_in[4];
  const float* out_bias = (const float*)d_in[5];
  const float* gamma    = (const float*)d_in[6];
  const unsigned char* mat_mask = (const unsigned char*)d_in[7]; // jnp.bool_ = 1 byte
  float* out = (float*)d_out;

  char* ws = (char*)d_ws;
  unsigned short* xb  = (unsigned short*)(ws);                    //  8 MB
  unsigned short* wtb = (unsigned short*)(ws + 8388608);          //  6 MB
  unsigned short* otb = (unsigned short*)(ws + 14680064);         //  2 MB
  unsigned short* Qb  = (unsigned short*)(ws + 16777216);         //  8 MB (B,H,L,HS)
  unsigned short* Kb  = (unsigned short*)(ws + 25165824);         //  8 MB (B,H,L,HS)
  unsigned short* Vt  = (unsigned short*)(ws + 33554432);         //  8 MB (B,H,HS,L)
  unsigned short* Ob  = (unsigned short*)(ws + 41943040);         //  8 MB (B*L, D)

  cvt_f32_bf16<<<(Mn * Dn) / 256, 256, 0, stream>>>(x, xb, Mn * Dn);
  transpose_cvt<<<(Dn * TDn) / 256, 256, 0, stream>>>(weights, wtb, Dn, TDn);
  transpose_cvt<<<(Dn * Dn) / 256, 256, 0, stream>>>(out_w, otb, Dn, Dn);

  qkv_gemm<<<(Mn / 32) * (TDn / 32) / 8, 256, 0, stream>>>(xb, wtb, in_bias, Qb, Kb, Vt);
  attn_kernel<<<(Bn * Hn * (Ln / 16)) / 8, 256, 0, stream>>>(Qb, Kb, Vt, adj, mat_mask, gamma, Ob);
  out_gemm<<<(Mn / 32) * (Dn / 32) / 8, 256, 0, stream>>>(Ob, otb, out_bias, out);
}
